// MultiHeadAttention_34806414967208
// MI455X (gfx1250) — matrix operand order repaired and checked by reading, not compiled, not run on hardware
//
#include <hip/hip_runtime.h>
#include <hip/hip_bf16.h>
#include <math.h>
#include <stdint.h>

// ---------------------------------------------------------------------------
// Types for CDNA5 WMMA (wave32): V_WMMA_F32_16X16X32_F16
// ---------------------------------------------------------------------------
typedef _Float16 h16;
typedef __attribute__((ext_vector_type(8)))  _Float16 v8h;
typedef __attribute__((ext_vector_type(16))) _Float16 v16h;
typedef __attribute__((ext_vector_type(8)))  float    v8f;
typedef __attribute__((ext_vector_type(4)))  unsigned int v4u;
typedef __attribute__((ext_vector_type(4)))  int v4i;
typedef __attribute__((ext_vector_type(8)))  int v8i;

__device__ __forceinline__ v8f wmma_f16(v16h a, v16h b, v8f c) {
    // (neg_a, A, neg_b, B, c_mod, C, reuse_a, reuse_b)
    return __builtin_amdgcn_wmma_f32_16x16x32_f16(false, a, false, b,
                                                  (short)0, c, false, false);
}

__device__ __forceinline__ h16 f2h(float f) { return (h16)f; }

// Build a 16-wide A/B fragment from two aligned 8-element chunks.
__device__ __forceinline__ v16h frag_from(const h16* p0, const h16* p1) {
    v8h a = *(const v8h*)p0;
    v8h b = *(const v8h*)p1;
    v16h r;
#pragma unroll
    for (int i = 0; i < 8; ++i) { r[i] = a[i]; r[8 + i] = b[i]; }
    return r;
}

// ---------------------------------------------------------------------------
// Tensor Data Mover: 2D tile (f16 elements) global -> LDS, per ISA D# layout.
// pad: +8 DWORDs every 16 DWORDs -> LDS row stride 96B (keeps 16B alignment,
// breaks the 4-way bank conflict of a 64B stride).
// ---------------------------------------------------------------------------
struct Tdm2D { v4u g0; v8i g1; };

__device__ __forceinline__ Tdm2D make_tdm_2d(unsigned lds_off, const void* gptr,
                                             unsigned tile_d0, unsigned tile_d1,
                                             unsigned long long stride_elems) {
    Tdm2D d;
    unsigned long long ga = (unsigned long long)(uintptr_t)gptr;
    d.g0[0] = 1u;                                       // count=1 (valid user D#)
    d.g0[1] = lds_off;                                  // LDS byte address
    d.g0[2] = (unsigned)ga;                             // global_addr[31:0]
    d.g0[3] = (unsigned)((ga >> 32) & 0x01FFFFFFull)    // global_addr[56:32]
            | 0x80000000u;                              // type=2 ("image")
    // ---- group1 (256 bits) ----
    unsigned long long q0 = (1ull << 16)                // data_size = 2 bytes
                          | (1ull << 20)                // pad_enable
                          | (3ull << 22)                // pad_interval: 16 DWORDs
                          | (7ull << 25)                // pad_amount: 8 DWORDs
                          | (0xFFFFull << 48);          // tensor_dim0[15:0] (huge: no OOB)
    unsigned long long q1 = 0x7FFFull                   // tensor_dim0[31:16]
                          | (0x7FFFFFFFull << 16)       // tensor_dim1 (huge: no OOB)
                          | ((unsigned long long)tile_d0 << 48);  // tile_dim0
    unsigned long long q2 = (unsigned long long)(tile_d1 & 0xFFFFu)   // tile_dim1
                          | ((stride_elems & 0xFFFFFFFFull) << 32);  // dim0_stride lo32
    unsigned long long q3 = (stride_elems >> 32) & 0xFFFFull;        // dim0_stride hi16
    d.g1[0] = (int)q0; d.g1[1] = (int)(q0 >> 32);
    d.g1[2] = (int)q1; d.g1[3] = (int)(q1 >> 32);
    d.g1[4] = (int)q2; d.g1[5] = (int)(q2 >> 32);
    d.g1[6] = (int)q3; d.g1[7] = (int)(q3 >> 32);
    return d;
}

__device__ __forceinline__ void tdm_issue(Tdm2D d) {
    v4i z = {0, 0, 0, 0};
#if __clang_major__ >= 23
    v8i z8 = {0, 0, 0, 0, 0, 0, 0, 0};
    __builtin_amdgcn_tensor_load_to_lds(d.g0, d.g1, z, z, z8, 0);
#else
    __builtin_amdgcn_tensor_load_to_lds(d.g0, d.g1, z, z, 0);
#endif
}

// ---------------------------------------------------------------------------
// fp32 -> f16 conversion
// ---------------------------------------------------------------------------
__global__ void cvt_f32_f16(const float* __restrict__ in, h16* __restrict__ out, int n) {
    int i = blockIdx.x * blockDim.x + threadIdx.x;
    if (i < n) out[i] = f2h(in[i]);
}

// ---------------------------------------------------------------------------
// C[M,N] (fp32) = A[M,K] (f16) @ W[N,K]^T (f16).  128x128 block tile,
// TDM double-buffered LDS staging, 8 waves each doing a 16x128 strip.
// Fragment loads are batched ahead of the WMMA chain so the 8 matrix ops can
// issue back-to-back behind a single dscnt wait instead of one wait per WMMA.
// ---------------------------------------------------------------------------
#define RS 48   // padded LDS row stride in f16 elements (96 bytes)

__global__ __launch_bounds__(256)
void gemm_xwt_tdm(const h16* __restrict__ A, const h16* __restrict__ W,
                  float* __restrict__ C, int M, int N, int K) {
    __shared__ __align__(16) h16 bufA[2][128 * RS];
    __shared__ __align__(16) h16 bufB[2][128 * RS];

    const int lane = threadIdx.x & 31;
    const int wave = threadIdx.x >> 5;
    const int g    = lane >> 4;
    const int lm   = lane & 15;

    const int nb128 = N >> 7;
    const int mb = blockIdx.x / nb128;
    const int nb = blockIdx.x % nb128;
    if (mb >= (M >> 7)) return;
    const int m0 = mb << 7;
    const int n0 = nb << 7;

    const unsigned ldsA[2] = { (unsigned)(uintptr_t)&bufA[0][0],
                               (unsigned)(uintptr_t)&bufA[1][0] };
    const unsigned ldsB[2] = { (unsigned)(uintptr_t)&bufB[0][0],
                               (unsigned)(uintptr_t)&bufB[1][0] };
    const unsigned long long Ks = (unsigned long long)K;
    const int nk = K >> 5;

    if (wave == 0) {   // prologue: tile 0 -> buffer 0
        tdm_issue(make_tdm_2d(ldsA[0], A + (size_t)m0 * K, 32, 128, Ks));
        tdm_issue(make_tdm_2d(ldsB[0], W + (size_t)n0 * K, 32, 128, Ks));
    }

    v8f acc[8] = {};
    for (int kb = 0; kb < nk; ++kb) {
        const int cur = kb & 1;
        if (wave == 0) {
            if (kb + 1 < nk) {   // prefetch next tile into the other buffer
                const h16* an = A + (size_t)m0 * K + (size_t)(kb + 1) * 32;
                const h16* bn = W + (size_t)n0 * K + (size_t)(kb + 1) * 32;
                tdm_issue(make_tdm_2d(ldsA[cur ^ 1], an, 32, 128, Ks));
                tdm_issue(make_tdm_2d(ldsB[cur ^ 1], bn, 32, 128, Ks));
                __builtin_amdgcn_s_wait_tensorcnt((short)2);  // tile kb landed
            } else {
                __builtin_amdgcn_s_wait_tensorcnt((short)0);
            }
        }
        __syncthreads();   // release tile kb to all waves

        // batch all fragment loads, then run the WMMA chain
        const h16* ap = &bufA[cur][(wave * 16 + lm) * RS];
        v16h af = frag_from(ap + g * 8, ap + 16 + g * 8);
        v16h bfr[8];
#pragma unroll
        for (int t = 0; t < 8; ++t) {
            const h16* bp = &bufB[cur][(t * 16 + lm) * RS + g * 8];
            bfr[t] = frag_from(bp, bp + 16);
        }
#pragma unroll
        for (int t = 0; t < 8; ++t)
            acc[t] = wmma_f16(af, bfr[t], acc[t]);

        __syncthreads();   // buffer may be refilled next iteration
    }

#pragma unroll
    for (int t = 0; t < 8; ++t)
#pragma unroll
        for (int r = 0; r < 8; ++r)
            C[(size_t)(m0 + wave * 16 + g * 8 + r) * N + n0 + t * 16 + lm] = acc[t][r];
}

// ---------------------------------------------------------------------------
// RoPE on fp32 Q,K (layout [B*S, D]) -> f16 [B,H,S,64]. hd=64 (32 pairs).
// ---------------------------------------------------------------------------
__global__ void rope_repack(const float* __restrict__ Qf, const float* __restrict__ Kf,
                            const int* __restrict__ pos,
                            h16* __restrict__ Qr, h16* __restrict__ Kr,
                            int B, int S, int H, int D, int n) {
    int tid = blockIdx.x * blockDim.x + threadIdx.x;
    if (tid >= n) return;
    int i = tid & 31;
    int h = (tid >> 5) % H;
    int s = (tid / (32 * H)) % S;
    int b = tid / (32 * H * S);

    float p   = (float)pos[s];
    float inv = __powf(10000.0f, -(float)(2 * i) / 64.0f);
    float ang = p * inv;
    float sn, cs;
    sincosf(ang, &sn, &cs);

    size_t src = (size_t)(b * S + s) * D + h * 64 + 2 * i;
    size_t dst = (size_t)((b * H + h) * S + s) * 64 + 2 * i;

    float qe = Qf[src], qo = Qf[src + 1];
    Qr[dst]     = f2h(qe * cs - qo * sn);
    Qr[dst + 1] = f2h(qe * sn + qo * cs);
    float ke = Kf[src], ko = Kf[src + 1];
    Kr[dst]     = f2h(ke * cs - ko * sn);
    Kr[dst + 1] = f2h(ke * sn + ko * cs);
}

// ---------------------------------------------------------------------------
// V fp32 [B*S, D] -> f16 transposed [B,H,64,S] (so PV B-frags are contiguous)
// ---------------------------------------------------------------------------
__global__ void repack_v(const float* __restrict__ Vf, h16* __restrict__ Vt,
                         int B, int S, int H, int D, int n) {
    int tid = blockIdx.x * blockDim.x + threadIdx.x;
    if (tid >= n) return;
    int d = tid & 63;
    int h = (tid >> 6) % H;
    int s = (tid / (64 * H)) % S;
    int b = tid / (64 * H * S);
    Vt[(size_t)((b * H + h) * 64 + d) * S + s] =
        f2h(Vf[(size_t)(b * S + s) * D + h * 64 + d]);
}

// ---------------------------------------------------------------------------
// Causal flash attention, hd = 64. One wave = 16 queries; key tiles of 32.
// Q,K: f16 [B,H,S,64]; Vt: f16 [B,H,64,S]; AO: fp32 [B,S,D].
// K/V fragment loads are batched ahead of their WMMA chains.
// ---------------------------------------------------------------------------
__global__ __launch_bounds__(256)
void flash_attn(const h16* __restrict__ Qr, const h16* __restrict__ Kr,
                const h16* __restrict__ Vt, float* __restrict__ AO,
                int S, int H, int D) {
    __shared__ __align__(16) h16 plds[8][16 * 32];   // P re-layout staging / wave

    const int lane = threadIdx.x & 31;
    const int wave = threadIdx.x >> 5;
    const int g    = lane >> 4;
    const int lm   = lane & 15;

    const int bh = blockIdx.y;
    const int b  = bh / H;
    const int h  = bh % H;
    const int q0 = (blockIdx.x * 8 + wave) * 16;

    const h16* Q  = Qr + (size_t)bh * S * 64;
    const h16* Kb = Kr + (size_t)bh * S * 64;
    const h16* V  = Vt + (size_t)bh * 64 * S;

    const h16* qrow = Q + (size_t)(q0 + lm) * 64;
    v16h qf[2];
#pragma unroll
    for (int f = 0; f < 2; ++f)
        qf[f] = frag_from(qrow + f * 32 + g * 8, qrow + f * 32 + 16 + g * 8);

    v8f o[4] = {};
    float mrow[8], lrow[8];
#pragma unroll
    for (int r = 0; r < 8; ++r) { mrow[r] = -1e30f; lrow[r] = 0.0f; }
    const float scale = 0.125f;   // 1/sqrt(64)
    const int qhi = q0 + 15;

    for (int j = 0; j <= qhi; j += 32) {
        __builtin_prefetch(Kb + (size_t)(j + 32 + lm) * 64, 0, 1);

        // ---- scores: batch 4 K fragments, then 4 WMMAs
        v16h kfr[4];
#pragma unroll
        for (int t = 0; t < 2; ++t)
#pragma unroll
            for (int f = 0; f < 2; ++f) {
                const h16* krow = Kb + (size_t)(j + t * 16 + lm) * 64 + f * 32 + g * 8;
                kfr[t * 2 + f] = frag_from(krow, krow + 16);
            }
        v8f sc[2] = {};
#pragma unroll
        for (int t = 0; t < 2; ++t)
#pragma unroll
            for (int f = 0; f < 2; ++f)
                sc[t] = wmma_f16(qf[f], kfr[t * 2 + f], sc[t]);

        // ---- online softmax (rows r + g*8; cols striped over lanes 0..15)
        float alpha[8];
#pragma unroll
        for (int r = 0; r < 8; ++r) {
            int   qg = q0 + g * 8 + r;
            float s0 = ((j + lm)      <= qg) ? sc[0][r] * scale : -1e30f;
            float s1 = ((j + 16 + lm) <= qg) ? sc[1][r] * scale : -1e30f;
            float tm = fmaxf(s0, s1);
            tm = fmaxf(tm, __shfl_xor(tm, 1, 32));
            tm = fmaxf(tm, __shfl_xor(tm, 2, 32));
            tm = fmaxf(tm, __shfl_xor(tm, 4, 32));
            tm = fmaxf(tm, __shfl_xor(tm, 8, 32));
            float mn = fmaxf(mrow[r], tm);
            float a  = __expf(mrow[r] - mn);
            float p0 = __expf(s0 - mn);
            float p1 = __expf(s1 - mn);
            float rs = p0 + p1;
            rs += __shfl_xor(rs, 1, 32);
            rs += __shfl_xor(rs, 2, 32);
            rs += __shfl_xor(rs, 4, 32);
            rs += __shfl_xor(rs, 8, 32);
            mrow[r]  = mn;
            lrow[r]  = lrow[r] * a + rs;
            alpha[r] = a;
            int prow = g * 8 + r;
            plds[wave][prow * 32 + lm]      = f2h(p0);
            plds[wave][prow * 32 + 16 + lm] = f2h(p1);
        }
#pragma unroll
        for (int t = 0; t < 4; ++t)
#pragma unroll
            for (int r = 0; r < 8; ++r) o[t][r] *= alpha[r];

        asm volatile("s_wait_dscnt 0" ::: "memory");

        // ---- P@V: batch P + 4 V fragments, then 4 WMMAs
        const h16* pb = &plds[wave][lm * 32 + g * 8];
        v16h pf = frag_from(pb, pb + 16);
        v16h vfr[4];
#pragma unroll
        for (int t = 0; t < 4; ++t) {
            const h16* vrow = V + (size_t)(t * 16 + lm) * S + j + g * 8;
            vfr[t] = frag_from(vrow, vrow + 16);
        }
#pragma unroll
        for (int t = 0; t < 4; ++t)
            o[t] = wmma_f16(pf, vfr[t], o[t]);
    }

#pragma unroll
    for (int r = 0; r < 8; ++r) {
        int   qg  = q0 + g * 8 + r;
        float inv = 1.0f / lrow[r];
#pragma unroll
        for (int t = 0; t < 4; ++t)
            AO[(size_t)(b * S + qg) * D + h * 64 + t * 16 + lm] = o[t][r] * inv;
    }
}

// ---------------------------------------------------------------------------
// Host orchestration
// ---------------------------------------------------------------------------
extern "C" void kernel_launch(void* const* d_in, const int* in_sizes, int n_in,
                              void* d_out, int out_size, void* d_ws, size_t ws_size,
                              hipStream_t stream) {
    (void)n_in; (void)out_size; (void)ws_size;
    const float* x   = (const float*)d_in[0];
    const int*   pos = (const int*)d_in[1];
    const float* Wq  = (const float*)d_in[2];
    const float* Wk  = (const float*)d_in[3];
    const float* Wv  = (const float*)d_in[4];
    const float* Wo  = (const float*)d_in[5];

    const int S  = in_sizes[1];
    const int D  = (int)llround(sqrt((double)in_sizes[2]));
    const int BS = in_sizes[0] / D;
    const int B  = BS / S;
    const int H  = 16;

    const size_t e  = (size_t)BS * D;
    const size_t dd = (size_t)D * D;

    char* w = (char*)d_ws;
    size_t off = 0;
    auto up = [](size_t v) { return (v + 255) & ~(size_t)255; };
    h16*   xb  = (h16*)(w + off);   off = up(off + e * 2);
    h16*   wqb = (h16*)(w + off);   off = up(off + dd * 2);
    h16*   wkb = (h16*)(w + off);   off = up(off + dd * 2);
    h16*   wvb = (h16*)(w + off);   off = up(off + dd * 2);
    h16*   wob = (h16*)(w + off);   off = up(off + dd * 2);
    float* qf  = (float*)(w + off); off = up(off + e * 4);
    float* kf  = (float*)(w + off); off = up(off + e * 4);
    float* vf  = (float*)(w + off); off = up(off + e * 4);
    h16*   qr  = (h16*)(w + off);   off = up(off + e * 2);
    h16*   kr  = (h16*)(w + off);   off = up(off + e * 2);
    h16*   vt  = (h16*)(w + off);   off = up(off + e * 2);
    float* ao  = qf;                // Q fp32 dead after rope_repack
    h16*   aob = (h16*)kf;          // K fp32 dead after rope_repack

    const int T = 256;
    cvt_f32_f16<<<((int)e + T - 1) / T, T, 0, stream>>>(x,  xb,  (int)e);
    cvt_f32_f16<<<((int)dd + T - 1) / T, T, 0, stream>>>(Wq, wqb, (int)dd);
    cvt_f32_f16<<<((int)dd + T - 1) / T, T, 0, stream>>>(Wk, wkb, (int)dd);
    cvt_f32_f16<<<((int)dd + T - 1) / T, T, 0, stream>>>(Wv, wvb, (int)dd);
    cvt_f32_f16<<<((int)dd + T - 1) / T, T, 0, stream>>>(Wo, wob, (int)dd);

    const int gblk = (BS / 128) * (D / 128);
    gemm_xwt_tdm<<<gblk, T, 0, stream>>>(xb, wqb, qf, BS, D, D);
    gemm_xwt_tdm<<<gblk, T, 0, stream>>>(xb, wkb, kf, BS, D, D);
    gemm_xwt_tdm<<<gblk, T, 0, stream>>>(xb, wvb, vf, BS, D, D);

    const int nrope = B * S * H * 32;
    rope_repack<<<(nrope + T - 1) / T, T, 0, stream>>>(qf, kf, pos, qr, kr, B, S, H, D, nrope);
    repack_v<<<((int)e + T - 1) / T, T, 0, stream>>>(vf, vt, B, S, H, D, (int)e);

    flash_attn<<<dim3(S / 128, B * H), T, 0, stream>>>(qr, kr, vt, ao, S, H, D);

    cvt_f32_f16<<<((int)e + T - 1) / T, T, 0, stream>>>(ao, aob, (int)e);
    gemm_xwt_tdm<<<gblk, T, 0, stream>>>(aob, wob, (float*)d_out, BS, D, D);
}
